// DANetHead_37924561223996
// MI455X (gfx1250) — matrix operand order repaired and checked by reading, not compiled, not run on hardware
//
#include <hip/hip_runtime.h>

// ---------------------------------------------------------------------------
// DANet head for MI455X (gfx1250, wave32, WMMA bf16 16x16x32 f32-acc,
// Tensor Data Mover staging for the conv implicit-GEMM)
// ---------------------------------------------------------------------------

typedef __attribute__((ext_vector_type(16))) __bf16 v16bf;
typedef __attribute__((ext_vector_type(8)))  __bf16 v8bf;
typedef __attribute__((ext_vector_type(8)))  float  v8f;
typedef __attribute__((ext_vector_type(4)))  unsigned int u32x4;
typedef __attribute__((ext_vector_type(8)))  int i32x8;
typedef __attribute__((ext_vector_type(4)))  int i32x4;

#define B_     2
#define CIN    256
#define CD     128
#define HW     64
#define NPIX   4096       // 64*64
#define KOUT   19
#define GROUPS 32

__device__ inline __bf16 to_bf16(float f) {
    unsigned u = __float_as_uint(f);
    unsigned r = (u + 0x7FFFu + ((u >> 16) & 1u)) >> 16;   // RNE
    unsigned short h = (unsigned short)r;
    return __builtin_bit_cast(__bf16, h);
}

__device__ inline v8f wmma_bf16(v16bf a, v16bf b, v8f c) {
    return __builtin_amdgcn_wmma_f32_16x16x32_bf16(
        false, a, false, b, (short)0, c, false, false);
}

// A fragment 16x32 (row-major source, row stride in elements).
// lane&15 = row M ; chunks K=[kb,kb+8) and K=[kb+16,kb+24), kb=(lane>>4)*8
__device__ inline v16bf load_frag_a(const __bf16* base, int row_stride, int lane) {
    const __bf16* p = base + (lane & 15) * row_stride + ((lane >> 4) << 3);
    union { v16bf v; v8bf h[2]; } u;
    u.h[0] = *(const v8bf*)(p);
    u.h[1] = *(const v8bf*)(p + 16);
    return u.v;
}

// B fragment 32x16 (column-major source: col_stride between columns).
// lane&15 = col N ; chunks K=[kb,kb+8) and K=[kb+16,kb+24), kb=(lane>>4)*8
__device__ inline v16bf load_frag_b(const __bf16* base, int col_stride, int lane) {
    const __bf16* p = base + (lane & 15) * col_stride + ((lane >> 4) << 3);
    union { v16bf v; v8bf h[2]; } u;
    u.h[0] = *(const v8bf*)(p);
    u.h[1] = *(const v8bf*)(p + 16);
    return u.v;
}

// wave-local LDS sync (producer lanes -> consumer lanes within one wave)
__device__ inline void lds_wave_sync() {
    __builtin_amdgcn_wave_barrier();
    asm volatile("s_wait_dscnt 0" ::: "memory");
    __builtin_amdgcn_wave_barrier();
}

// low 32 bits of a generic pointer to LDS == the DS/TDM byte address
__device__ inline unsigned lds_off(const void* p) {
    return (unsigned)(unsigned long long)(uintptr_t)p;
}

// TDM: DMA a 2D tile [64 rows x 32 bf16 elems] from global into LDS.
// Global: row stride = 256 elements (NHWC channel stride); rows contiguous 64B.
// LDS: row stride 80B via padding (pad_interval 16 dwords, pad_amount 4 dwords).
__device__ inline void tdm_load_tile(unsigned lds_addr, const void* gptr) {
    unsigned long long ga = (unsigned long long)(uintptr_t)gptr;
    u32x4 g0;
    g0[0] = 1u;                                        // count=1 (valid), user mode
    g0[1] = lds_addr;                                  // lds_addr [63:32]
    g0[2] = (unsigned)(ga & 0xFFFFFFFFu);              // global_addr lo
    g0[3] = (unsigned)((ga >> 32) & 0x01FFFFFFu)       // global_addr hi (57-bit)
          | (2u << 30);                                // type = 2 (image)
    i32x8 g1;
    // wg_mask=0 | data_size=1 (2B) | pad_enable | pad_interval=3 (16 dw) | pad_amount=3 (4 dw)
    g1[0] = (int)((1u << 16) | (1u << 20) | (3u << 22) | (3u << 25));
    g1[1] = (int)((32u & 0xFFFFu) << 16);              // barrier_addr=0 | tensor_dim0 lo16 (=32)
    g1[2] = (int)((32u >> 16) | (64u << 16));          // tensor_dim0 hi | tensor_dim1 lo16 (=64)
    g1[3] = (int)((64u >> 16) | (32u << 16));          // tensor_dim1 hi | tile_dim0 (=32)
    g1[4] = (int)(64u);                                // tile_dim1 (=64) | tile_dim2=0
    g1[5] = (int)(256u);                               // tensor_dim0_stride lo32 (=256 elems)
    g1[6] = 0;                                         // stride0 hi16 | stride1 lo16
    g1[7] = 0;
    i32x4 gz4 = {0, 0, 0, 0};
    i32x8 gz8 = {0, 0, 0, 0, 0, 0, 0, 0};
    __builtin_amdgcn_tensor_load_to_lds(g0, g1, gz4, gz4, gz8, 0);
}

// ---------------------------------------------------------------------------
// 0) conversions
// ---------------------------------------------------------------------------
// x [B][C][H][W] f32  ->  xT [B][H][W][C] bf16  (NHWC so TDM tiles are contiguous)
__global__ void cvt_x_nhwc(const float* __restrict__ x, __bf16* __restrict__ xT, int n) {
    int i = blockIdx.x * 256 + threadIdx.x;
    if (i >= n) return;
    int px = i & (NPIX - 1);
    int c  = (i >> 12) & (CIN - 1);
    int b  = i >> 20;
    xT[((size_t)b * NPIX + px) * CIN + c] = to_bf16(x[i]);
}

// dest layout: [head][dy][dx][O=128][Ci=256]; src [O][Ci][3][3]
__global__ void cvt_w_bf16(const float* __restrict__ wp, const float* __restrict__ wc,
                           __bf16* __restrict__ wbf) {
    int i = blockIdx.x * 256 + threadIdx.x;
    const int n = 2 * 9 * CD * CIN;
    if (i >= n) return;
    int ci  = i & (CIN - 1);
    int o   = (i >> 8) & (CD - 1);
    int tap = (i / (CIN * CD)) % 9;
    int h   = i / (CIN * CD * 9);
    const float* src = h ? wc : wp;
    wbf[i] = to_bf16(src[(o * CIN + ci) * 9 + tap]);
}

// ---------------------------------------------------------------------------
// 1) conv3x3 (implicit GEMM, 9-tap accumulation) + GN partial sums
//    grid: head(2) x b(2) x y(64); block 256 = 8 waves, wave = 16 out chans
//    B tiles staged by TDM (double buffered, TENSORcnt pipelined)
// ---------------------------------------------------------------------------
__global__ __launch_bounds__(256) void conv3x3_wmma(
    const __bf16* __restrict__ xT,    // [B][64][64][CIN] bf16 (NHWC)
    const __bf16* __restrict__ wbf,   // [head][3][3][CD][CIN]
    float* __restrict__ feat,         // [head][B][CD][NPIX]
    float* __restrict__ partial)      // [head][B][y=64][G=32][2]
{
    int wg = blockIdx.x;
    int y  = wg & 63;
    int b  = (wg >> 6) & 1;
    int h  = wg >> 7;
    int tid  = threadIdx.x;
    int wave = tid >> 5, lane = tid & 31;
    int cbase = wave * 16;

    __shared__ __bf16 Bs[2][66 * 40];   // [px = x'+1][32 ch], row stride 40 halfs = 80B

    // halo columns (px=0, px=65) are statically OOB -> zero them once
    for (int i = tid; i < 2 * 2 * 32; i += 256) {
        int c = i & 31;
        int px = ((i >> 5) & 1) ? 65 : 0;
        int bufi = i >> 6;
        Bs[bufi][px * 40 + c] = __builtin_bit_cast(__bf16, (unsigned short)0);
    }
    __syncthreads();

    v8f acc[4] = {};
    for (int dy = 0; dy < 3; ++dy) {
        int ys = y + dy - 1;
        if ((unsigned)ys >= 64u) continue;                       // uniform over WG
        const __bf16* grow = xT + ((size_t)(b * 64 + ys) * 64) * CIN;  // (x=0, c=0)
        int buf = 0;
        if (wave == 0)
            tdm_load_tile(lds_off(&Bs[0][40]), grow);            // px=1 row, ck=0
        for (int ck = 0; ck < CIN; ck += 32) {
            if (wave == 0) {
                if (ck + 32 < CIN) {
                    tdm_load_tile(lds_off(&Bs[buf ^ 1][40]), grow + (ck + 32));
                    __builtin_amdgcn_s_wait_tensorcnt(1);        // current buf done
                } else {
                    __builtin_amdgcn_s_wait_tensorcnt(0);
                }
            }
            __syncthreads();
            const __bf16* Bbuf = &Bs[buf][0];
            for (int dx = 0; dx < 3; ++dx) {
                const __bf16* wptr = wbf + ((size_t)(((h * 3 + dy) * 3 + dx) * CD + cbase)) * CIN + ck;
                v16bf afr = load_frag_a(wptr, CIN, lane);
                for (int t = 0; t < 4; ++t) {
                    v16bf bfr = load_frag_b(Bbuf + (t * 16 + dx) * 40, 40, lane);
                    acc[t] = wmma_bf16(afr, bfr, acc[t]);
                }
            }
            __syncthreads();                                     // WAR on buf
            buf ^= 1;
        }
    }

    // store conv output + deterministic GN partials
    float sum0 = 0.f, sum1 = 0.f, ssq0 = 0.f, ssq1 = 0.f;
    int half = lane >> 4;
    for (int t = 0; t < 4; ++t)
        for (int v = 0; v < 8; ++v) {
            float val = acc[t][v];
            int c  = cbase + v + (half << 3);
            int px = t * 16 + (lane & 15);
            feat[((size_t)((h * 2 + b) * CD + c)) * NPIX + y * 64 + px] = val;
            if (v < 4) { sum0 += val; ssq0 += val * val; }
            else       { sum1 += val; ssq1 += val * val; }
        }
    for (int m = 1; m < 16; m <<= 1) {
        sum0 += __shfl_xor(sum0, m, 32);
        sum1 += __shfl_xor(sum1, m, 32);
        ssq0 += __shfl_xor(ssq0, m, 32);
        ssq1 += __shfl_xor(ssq1, m, 32);
    }
    if ((lane & 15) == 0) {
        int g0 = wave * 4 + half * 2;
        float* p = partial + ((size_t)((h * 2 + b) * 64 + y)) * 64;
        p[(g0 + 0) * 2 + 0] = sum0;  p[(g0 + 0) * 2 + 1] = ssq0;
        p[(g0 + 1) * 2 + 0] = sum1;  p[(g0 + 1) * 2 + 1] = ssq1;
    }
}

// ---------------------------------------------------------------------------
// 2) GN finalize: per-channel affine a,b   (512 threads)
// ---------------------------------------------------------------------------
__global__ void gn_finalize(const float* __restrict__ partial,
                            const float* __restrict__ s_pam, const float* __restrict__ b_pam,
                            const float* __restrict__ s_cam, const float* __restrict__ b_cam,
                            float* __restrict__ gnab) {
    int i = blockIdx.x * 256 + threadIdx.x;      // [head][b][c]
    if (i >= 2 * 2 * CD) return;
    int c = i & (CD - 1), hb = i >> 7;
    int g = c >> 2;
    float s = 0.f, sq = 0.f;
    for (int y = 0; y < 64; ++y) {
        const float* p = partial + ((size_t)(hb * 64 + y)) * 64 + g * 2;
        s += p[0]; sq += p[1];
    }
    const float inv = 1.f / 16384.f;             // 4 ch * 4096 px
    float mean = s * inv;
    float var  = sq * inv - mean * mean;
    float rstd = rsqrtf(var + 1e-5f);
    float sc = (hb >> 1) ? s_cam[c] : s_pam[c];
    float bi = (hb >> 1) ? b_cam[c] : b_pam[c];
    float a = sc * rstd;
    gnab[i * 2 + 0] = a;
    gnab[i * 2 + 1] = bi - mean * a;
}

// ---------------------------------------------------------------------------
// 3) normalize + relu; emit fp32 feat (in place), bf16 feat [C][N], bf16 featT [N][C]
// ---------------------------------------------------------------------------
__global__ __launch_bounds__(256) void gn_relu_pack(
    float* __restrict__ feat, const float* __restrict__ gnab,
    __bf16* __restrict__ featbf, __bf16* __restrict__ featT) {
    int i = blockIdx.x * 256 + threadIdx.x;      // 2*2*128*4096
    int n  = i & (NPIX - 1);
    int c  = (i >> 12) & (CD - 1);
    int hb = i >> 19;
    float a  = gnab[(hb * CD + c) * 2 + 0];
    float bb = gnab[(hb * CD + c) * 2 + 1];
    float r = fmaxf(feat[i] * a + bb, 0.f);
    feat[i] = r;
    __bf16 rb = to_bf16(r);
    featbf[i] = rb;
    featT[((size_t)hb << 19) + ((size_t)n << 7) + c] = rb;
}

// ---------------------------------------------------------------------------
// 4) CAM gram: G[b][c][d] = sum_n F[c][n] F[d][n]   (WMMA, K=4096)
// ---------------------------------------------------------------------------
__global__ __launch_bounds__(256) void cam_gram_wmma(const __bf16* __restrict__ featbf,
                                                     float* __restrict__ gram) {
    int b = blockIdx.x;
    int wave = threadIdx.x >> 5, lane = threadIdx.x & 31;
    const __bf16* F = featbf + ((size_t)(2 + b)) * CD * NPIX;   // cam head
    int rbase = wave * 16;
    v8f acc[8] = {};
    for (int kk = 0; kk < NPIX; kk += 32) {
        v16bf afr = load_frag_a(F + (size_t)rbase * NPIX + kk, NPIX, lane);
        for (int t = 0; t < 8; ++t) {
            v16bf bfr = load_frag_b(F + (size_t)(t * 16) * NPIX + kk, NPIX, lane);
            acc[t] = wmma_bf16(afr, bfr, acc[t]);
        }
    }
    for (int t = 0; t < 8; ++t)
        for (int v = 0; v < 8; ++v) {
            int c = rbase + v + ((lane >> 4) << 3);
            int d = t * 16 + (lane & 15);
            gram[((size_t)(b * CD + c)) * CD + d] = acc[t][v];
        }
}

// 4b) row softmax over 128 (256 serial rows — negligible work)
__global__ void cam_softmax(const float* __restrict__ gram, __bf16* __restrict__ camw) {
    int r = blockIdx.x * 256 + threadIdx.x;
    if (r >= 2 * CD) return;
    const float* row = gram + (size_t)r * CD;
    float m = row[0];
    for (int j = 1; j < CD; ++j) m = fmaxf(m, row[j]);
    float s = 0.f;
    for (int j = 0; j < CD; ++j) s += __expf(row[j] - m);
    float inv = 1.f / s;
    for (int j = 0; j < CD; ++j) camw[(size_t)r * CD + j] = to_bf16(__expf(row[j] - m) * inv);
}

// 4c) weighted_cam[c][n] = sum_d camw[c][d] * F[d][n]
__global__ __launch_bounds__(256) void cam_apply_wmma(const __bf16* __restrict__ camw,
                                                      const __bf16* __restrict__ featT,
                                                      float* __restrict__ wcam) {
    int wg = blockIdx.x;                 // b(2) x pxtile(64)
    int pt = wg & 63, b = wg >> 6;
    int wave = threadIdx.x >> 5, lane = threadIdx.x & 31;
    int cbase = wave * 16;
    const __bf16* A = camw + (size_t)b * CD * CD;
    const __bf16* T = featT + ((size_t)(2 + b)) * NPIX * CD;    // cam head
    int px0 = pt * 64;
    v8f acc[4] = {};
    for (int kk = 0; kk < CD; kk += 32) {
        v16bf afr = load_frag_a(A + (size_t)cbase * CD + kk, CD, lane);
        for (int t = 0; t < 4; ++t) {
            v16bf bfr = load_frag_b(T + (size_t)(px0 + t * 16) * CD + kk, CD, lane);
            acc[t] = wmma_bf16(afr, bfr, acc[t]);
        }
    }
    for (int t = 0; t < 4; ++t)
        for (int v = 0; v < 8; ++v) {
            int c  = cbase + v + ((lane >> 4) << 3);
            int px = px0 + t * 16 + (lane & 15);
            wcam[((size_t)(b * CD + c)) * NPIX + px] = acc[t][v];
        }
}

// ---------------------------------------------------------------------------
// 5) PAM flash attention: O = softmax(Q K^T) V, Q=K=V = pam features
//    grid: b(2) x qtile(64) ; block 128 = 4 waves, wave = 16 query rows
// ---------------------------------------------------------------------------
__global__ __launch_bounds__(128) void pam_flash_wmma(const __bf16* __restrict__ featT,
                                                      const __bf16* __restrict__ featbf,
                                                      float* __restrict__ opam) {
    int wg = blockIdx.x;
    int qt = wg & 63, b = wg >> 6;
    int wave = threadIdx.x >> 5, lane = threadIdx.x & 31;
    const __bf16* T = featT + ((size_t)b) * NPIX * CD;   // pam head [N][C]
    const __bf16* F = featbf + ((size_t)b) * CD * NPIX;  // pam head [C][N]
    int q0 = qt * 64 + wave * 16;

    __shared__ __bf16 Pbuf[4][16 * 72];                  // per-wave P tile (stride 72)
    __bf16* Pb = &Pbuf[wave][0];

    v8f oacc[8] = {};
    float mrow[8], lrow[8];
    for (int v = 0; v < 8; ++v) { mrow[v] = -INFINITY; lrow[v] = 0.f; }

    for (int j0 = 0; j0 < NPIX; j0 += 64) {
        // prefetch next K tile (16KB of featT) while computing this one
        if (j0 + 64 < NPIX) {
            const __bf16* nt = T + (size_t)(j0 + 64) * CD;
            for (int pf = 0; pf < 8; ++pf)
                __builtin_prefetch(nt + ((size_t)(pf * 32 + lane)) * 32, 0, 0);
        }
        // S tile (16 x 64), K=128
        v8f s[4] = {};
        for (int kc = 0; kc < 4; ++kc) {
            v16bf afr = load_frag_a(T + (size_t)q0 * CD + kc * 32, CD, lane);
            for (int t = 0; t < 4; ++t) {
                v16bf bfr = load_frag_b(T + (size_t)(j0 + t * 16) * CD + kc * 32, CD, lane);
                s[t] = wmma_bf16(afr, bfr, s[t]);
            }
        }
        // online softmax, rows live at (vgpr v, lane half); reduce over lane bits 0..3
        float scale[8];
        for (int v = 0; v < 8; ++v) {
            float mx = fmaxf(fmaxf(s[0][v], s[1][v]), fmaxf(s[2][v], s[3][v]));
            for (int mb = 1; mb < 16; mb <<= 1) mx = fmaxf(mx, __shfl_xor(mx, mb, 32));
            float nm = fmaxf(mrow[v], mx);
            scale[v] = __expf(mrow[v] - nm);
            float rs = 0.f;
            for (int t = 0; t < 4; ++t) {
                float p = __expf(s[t][v] - nm);
                s[t][v] = p;
                rs += p;
            }
            for (int mb = 1; mb < 16; mb <<= 1) rs += __shfl_xor(rs, mb, 32);
            lrow[v] = lrow[v] * scale[v] + rs;
            mrow[v] = nm;
        }
        for (int t = 0; t < 8; ++t)
            for (int v = 0; v < 8; ++v)
                oacc[t][v] *= scale[v];
        // P (D layout) -> LDS -> A fragments
        for (int t = 0; t < 4; ++t)
            for (int v = 0; v < 8; ++v)
                Pb[(v + ((lane >> 4) << 3)) * 72 + t * 16 + (lane & 15)] = to_bf16(s[t][v]);
        lds_wave_sync();
        // O += P V : K=64 (2 chunks), 8 channel tiles
        for (int kc = 0; kc < 2; ++kc) {
            v16bf pfr = load_frag_a(Pb + kc * 32, 72, lane);
            for (int t = 0; t < 8; ++t) {
                v16bf vfr = load_frag_b(F + (size_t)(t * 16) * NPIX + (j0 + kc * 32), NPIX, lane);
                oacc[t] = wmma_bf16(pfr, vfr, oacc[t]);
            }
        }
        lds_wave_sync();   // WAR: next iteration overwrites Pb
    }
    // finalize: O[q][c] / l[q]  ->  opam[b][c][q]
    for (int t = 0; t < 8; ++t)
        for (int v = 0; v < 8; ++v) {
            int c = t * 16 + (lane & 15);
            int q = q0 + v + ((lane >> 4) << 3);
            opam[((size_t)(b * CD + c)) * NPIX + q] = oacc[t][v] / lrow[v];
        }
}

// ---------------------------------------------------------------------------
// 6) fuse residual streams + 1x1 predictor (K=19)
// ---------------------------------------------------------------------------
__global__ __launch_bounds__(256) void fuse_pred(
    const float* __restrict__ feat, const float* __restrict__ opam,
    const float* __restrict__ wcam, const float* __restrict__ wpred,
    const float* __restrict__ bpred, float* __restrict__ logits) {
    __shared__ float Wp[KOUT * CD];
    for (int i = threadIdx.x; i < KOUT * CD; i += 256) Wp[i] = wpred[i];
    __syncthreads();
    int px = blockIdx.x * 256 + threadIdx.x;     // 2*4096
    int n = px & (NPIX - 1), b = px >> 12;
    const float* fp = feat + ((size_t)b) * CD * NPIX + n;         // pam head
    const float* fc = feat + ((size_t)(2 + b)) * CD * NPIX + n;   // cam head
    const float* op = opam + ((size_t)b) * CD * NPIX + n;
    const float* wc = wcam + ((size_t)b) * CD * NPIX + n;
    float acc[KOUT];
    for (int k = 0; k < KOUT; ++k) acc[k] = bpred[k];
    for (int c = 0; c < CD; ++c) {
        float f = fp[(size_t)c * NPIX] + op[(size_t)c * NPIX]
                + fc[(size_t)c * NPIX] + wc[(size_t)c * NPIX];
        for (int k = 0; k < KOUT; ++k) acc[k] = fmaf(Wp[k * CD + c], f, acc[k]);
    }
    for (int k = 0; k < KOUT; ++k)
        logits[((size_t)(b * KOUT + k)) * NPIX + n] = acc[k];
}

// ---------------------------------------------------------------------------
// 7) bilinear 4x upsample, half-pixel centers
// ---------------------------------------------------------------------------
__global__ void upsample4x(const float* __restrict__ logits, float* __restrict__ out, int n) {
    int i = blockIdx.x * 256 + threadIdx.x;
    if (i >= n) return;
    int xo = i & 255, yo = (i >> 8) & 255, bk = i >> 16;
    float sx = (xo + 0.5f) * 0.25f - 0.5f;
    float sy = (yo + 0.5f) * 0.25f - 0.5f;
    int x0 = (int)floorf(sx); float fx = sx - (float)x0;
    int y0 = (int)floorf(sy); float fy = sy - (float)y0;
    int x1 = min(max(x0 + 1, 0), 63); x0 = min(max(x0, 0), 63);
    int y1 = min(max(y0 + 1, 0), 63); y0 = min(max(y0, 0), 63);
    const float* L = logits + (size_t)bk * NPIX;
    float v00 = L[y0 * 64 + x0], v01 = L[y0 * 64 + x1];
    float v10 = L[y1 * 64 + x0], v11 = L[y1 * 64 + x1];
    out[i] = (v00 * (1.f - fx) + v01 * fx) * (1.f - fy)
           + (v10 * (1.f - fx) + v11 * fx) * fy;
}

// ---------------------------------------------------------------------------
extern "C" void kernel_launch(void* const* d_in, const int* in_sizes, int n_in,
                              void* d_out, int out_size, void* d_ws, size_t ws_size,
                              hipStream_t stream) {
    (void)in_sizes; (void)n_in; (void)out_size; (void)ws_size;
    const float* x      = (const float*)d_in[0];
    const float* w_pam  = (const float*)d_in[1];
    const float* s_pam  = (const float*)d_in[2];
    const float* b_pam  = (const float*)d_in[3];
    const float* w_cam  = (const float*)d_in[4];
    const float* s_cam  = (const float*)d_in[5];
    const float* b_cam  = (const float*)d_in[6];
    const float* w_pred = (const float*)d_in[7];
    const float* b_pred = (const float*)d_in[8];
    float* out = (float*)d_out;

    char* ws = (char*)d_ws;
    size_t off = 0;
    auto alloc = [&](size_t bytes) -> void* {
        void* p = ws + off;
        off += (bytes + 255) & ~(size_t)255;
        return p;
    };
    __bf16* xT      = (__bf16*)alloc((size_t)B_ * CIN * NPIX * 2);     // NHWC
    __bf16* wbf     = (__bf16*)alloc((size_t)2 * 9 * CD * CIN * 2);
    float*  feat    = (float*) alloc((size_t)2 * B_ * CD * NPIX * 4);
    __bf16* featbf  = (__bf16*)alloc((size_t)2 * B_ * CD * NPIX * 2);
    __bf16* featT   = (__bf16*)alloc((size_t)2 * B_ * NPIX * CD * 2);
    float*  partial = (float*) alloc((size_t)2 * B_ * 64 * 64 * 4);
    float*  gnab    = (float*) alloc((size_t)2 * B_ * CD * 2 * 4);
    float*  gram    = (float*) alloc((size_t)B_ * CD * CD * 4);
    __bf16* camw    = (__bf16*)alloc((size_t)B_ * CD * CD * 2);
    float*  wcam    = (float*) alloc((size_t)B_ * CD * NPIX * 4);
    float*  opam    = (float*) alloc((size_t)B_ * CD * NPIX * 4);
    float*  logits  = (float*) alloc((size_t)B_ * KOUT * NPIX * 4);

    const int nx = B_ * CIN * NPIX;
    cvt_x_nhwc<<<(nx + 255) / 256, 256, 0, stream>>>(x, xT, nx);
    const int nw = 2 * 9 * CD * CIN;
    cvt_w_bf16<<<(nw + 255) / 256, 256, 0, stream>>>(w_pam, w_cam, wbf);

    conv3x3_wmma<<<2 * B_ * 64, 256, 0, stream>>>(xT, wbf, feat, partial);
    gn_finalize<<<2, 256, 0, stream>>>(partial, s_pam, b_pam, s_cam, b_cam, gnab);
    gn_relu_pack<<<(2 * B_ * CD * NPIX) / 256, 256, 0, stream>>>(feat, gnab, featbf, featT);

    cam_gram_wmma<<<B_, 256, 0, stream>>>(featbf, gram);
    cam_softmax<<<1, 256, 0, stream>>>(gram, camw);
    cam_apply_wmma<<<B_ * 64, 256, 0, stream>>>(camw, featT, wcam);

    pam_flash_wmma<<<B_ * 64, 128, 0, stream>>>(featT, featbf, opam);

    fuse_pred<<<(B_ * NPIX) / 256, 256, 0, stream>>>(feat, opam, wcam, w_pred, b_pred, logits);

    const int nout = B_ * KOUT * 256 * 256;
    upsample4x<<<(nout + 255) / 256, 256, 0, stream>>>(logits, out, nout);
}
